// MultiHeads_67233418052170
// MI455X (gfx1250) — compile-verified, matrix operand order checked by reading, not run on hardware
//
#include <hip/hip_runtime.h>
#include <hip/hip_bf16.h>
#include <math.h>

// ---------------------------------------------------------------------------
// MultiHeads cross-attention pipeline for MI455X (gfx1250), bf16 WMMA + TDM.
//   B=4, HEADS=8, D=128, HW=1024 (queries/head), N=2048 (keys/head)
// ---------------------------------------------------------------------------

#define SCALE_ATT 0.08838834764831845f   // 128^-0.5

typedef __attribute__((ext_vector_type(16))) __bf16 v16bf;
typedef __attribute__((ext_vector_type(8)))  float  v8f;
typedef __attribute__((ext_vector_type(4)))  unsigned int u32x4;
typedef __attribute__((ext_vector_type(4)))  int    i32x4;
typedef __attribute__((ext_vector_type(8)))  int    i32x8;

union Frag16 { v16bf v; unsigned int u[8]; };

// Load a 16x32 bf16 fragment (A-layout; identical pattern serves as B-layout
// when "row" = output column). ISA 16-bit layout: lanes 0-15 row=lane,
// VGPR j<4 -> K{2j,2j+1}, j>=4 -> K{16+...}; lanes 16-31 same rows, K +8/+24.
__device__ __forceinline__ v16bf load_frag_rm(const __bf16* base, int stride) {
    int lane = threadIdx.x & 31;
    int r  = lane & 15;
    int hi = lane >> 4;
    const __bf16* p = base + (size_t)r * stride + hi * 8;
    Frag16 f;
#pragma unroll
    for (int j = 0; j < 4; ++j) f.u[j]     = *(const unsigned int*)(p + 2 * j);
#pragma unroll
    for (int j = 0; j < 4; ++j) f.u[4 + j] = *(const unsigned int*)(p + 16 + 2 * j);
    return f.v;
}

__device__ __forceinline__ v8f wmma_bf16f32(v16bf a, v16bf b, v8f c) {
    return __builtin_amdgcn_wmma_f32_16x16x32_bf16(false, a, false, b, (short)0, c,
                                                   false, false);
}

__device__ __forceinline__ float act_lrelu(float x) { return x >= 0.f ? x : 0.01f * x; }
__device__ __forceinline__ float act_gelu(float x) {
    return 0.5f * x * (1.f + erff(x * 0.70710678118654752f));
}

// ---------------------------------------------------------------------------
// Tensor Data Mover: 2D tile load Global -> LDS (bf16 elements).
// D# per CDNA5 ISA. tile0 = contiguous dim (elements),
// tile1 = rows, stride0 = row stride (elements), tensor dims for OOB.
// ---------------------------------------------------------------------------
__device__ __forceinline__ void tdm_load_2d(const __bf16* gsrc, __bf16* ldst,
                                            unsigned tile0, unsigned tile1,
                                            unsigned tdim0, unsigned tdim1,
                                            unsigned stride0) {
    unsigned long long ga = (unsigned long long)(uintptr_t)gsrc;
    unsigned lds_off = (unsigned)(uintptr_t)ldst;  // low 32 bits = LDS byte offset
    u32x4 g0;
    g0.x = 1u;                                     // count=1, user descriptor
    g0.y = lds_off;                                // lds_addr
    g0.z = (unsigned)(ga & 0xFFFFFFFFu);           // global_addr[31:0]
    g0.w = (unsigned)((ga >> 32) & 0x01FFFFFFu) | (2u << 30);  // [56:32] | type=2
    i32x8 g1;
    g1[0] = (int)(1u << 16);                       // wg_mask=0, data_size=1 (2B)
    g1[1] = (int)((tdim0 & 0xFFFFu) << 16);        // tensor_dim0[15:0]
    g1[2] = (int)(((tdim0 >> 16) & 0xFFFFu) | ((tdim1 & 0xFFFFu) << 16));
    g1[3] = (int)(((tdim1 >> 16) & 0xFFFFu) | ((tile0 & 0xFFFFu) << 16));
    g1[4] = (int)(tile1 & 0xFFFFu);                // tile_dim1 (tile_dim2=0)
    g1[5] = (int)stride0;                          // tensor_dim0_stride[31:0]
    g1[6] = 0;                                     // stride0 hi16 | stride1 lo16
    g1[7] = 0;
    i32x4 z4 = {0, 0, 0, 0};
#if defined(__clang_major__) && (__clang_major__ >= 23)
    i32x8 z8 = {0, 0, 0, 0, 0, 0, 0, 0};
    __builtin_amdgcn_tensor_load_to_lds(g0, g1, z4, z4, z8, 0);
#else
    __builtin_amdgcn_tensor_load_to_lds(g0, g1, z4, z4, 0);
#endif
}

// ---------------------------------------------------------------------------
// Elementwise helpers.
// ---------------------------------------------------------------------------
__global__ void cast_f32_bf16(const float* __restrict__ in, __bf16* __restrict__ out, int n) {
    int i = blockIdx.x * blockDim.x + threadIdx.x;
    if (i < n) out[i] = (__bf16)in[i];
}

__global__ void transpose_cast(const float* __restrict__ in, __bf16* __restrict__ out,
                               int C, int L) {
    int g = blockIdx.y;
    int i = blockIdx.x * blockDim.x + threadIdx.x;
    if (i >= C * L) return;
    int c = i / L, l = i - c * L;
    out[(size_t)g * C * L + (size_t)l * C + c] = (__bf16)in[(size_t)g * C * L + i];
}

// ---------------------------------------------------------------------------
// Generic WMMA GEMM: Y = act(X @ W^T + bias), 128 output cols.
// One wave computes a 16-row strip across 8 n-tiles; A-frags hoisted for K<=128.
// IN_MODE 0: X row-major [M,KIN]; 1: conv-q rows (bh,m) w/ per-head W slice.
// OUT_MODE 0: row-major; 1: per-head transposed (V^T); 2: head-interleaved.
// ---------------------------------------------------------------------------
template <int KIN, int ACT, int IN_MODE, int OUT_MODE, bool OUT_F32>
__global__ __launch_bounds__(256) void gemm_bias_act(
        const __bf16* __restrict__ X, const __bf16* __restrict__ W,
        const float* __restrict__ bias, void* __restrict__ Yv, int Mtiles) {
    int gw = (int)((blockIdx.x * blockDim.x + threadIdx.x) >> 5);
    if (gw >= Mtiles) return;
    int lane = threadIdx.x & 31;
    int r = lane & 15, hi = lane >> 4;
    int m0 = gw * 16;

    const __bf16* xbase;
    int xstride;
    const __bf16* wbase = W;
    const float*  bb    = bias;
    if (IN_MODE == 0) {
        xbase = X + (size_t)m0 * KIN;
        xstride = KIN;
    } else {  // rows (bh, m); strips never straddle heads (1024 % 16 == 0)
        int bh = m0 >> 10, m = m0 & 1023;
        xbase = X + ((size_t)(bh >> 3) * 1024 + m) * 128;
        xstride = 128;
        wbase = W + (size_t)(bh & 7) * 128 * KIN;
        bb    = bias + (bh & 7) * 128;
    }

    // Hoist A fragments when they fit (KIN<=128 -> at most 4 frags, 32 VGPRs).
    v16bf af[(KIN <= 128) ? (KIN / 32) : 1];
    if constexpr (KIN <= 128) {
#pragma unroll
        for (int ks = 0; ks < KIN / 32; ++ks)
            af[ks] = load_frag_rm(xbase + ks * 32, xstride);
    }

#pragma unroll
    for (int nt = 0; nt < 8; ++nt) {
        v8f acc;
#pragma unroll
        for (int j = 0; j < 8; ++j) acc[j] = 0.f;
        if constexpr (KIN <= 128) {
#pragma unroll
            for (int ks = 0; ks < KIN / 32; ++ks)
                acc = wmma_bf16f32(af[ks],
                        load_frag_rm(wbase + (size_t)nt * 16 * KIN + ks * 32, KIN), acc);
        } else {
            for (int k = 0; k < KIN; k += 32)
                acc = wmma_bf16f32(load_frag_rm(xbase + k, xstride),
                        load_frag_rm(wbase + (size_t)nt * 16 * KIN + k, KIN), acc);
        }
        int col = nt * 16 + r;
        float bv = bb[col];
#pragma unroll
        for (int j = 0; j < 8; ++j) {
            int row = m0 + j + hi * 8;  // C layout: VGPR j -> M=j (+8 for hi half)
            float v = acc[j] + bv;
            if (ACT == 1)      v = act_lrelu(v);
            else if (ACT == 2) v = act_gelu(v);
            size_t idx;
            if (OUT_MODE == 0) {
                idx = (size_t)row * 128 + col;
            } else if (OUT_MODE == 1) {
                int bh = row >> 11, n = row & 2047;
                idx = (size_t)bh * 128 * 2048 + (size_t)col * 2048 + n;
            } else {
                int bh = row >> 10, m = row & 1023;
                int b_ = bh >> 3, h_ = bh & 7;
                idx = ((size_t)(b_ * 1024 + m)) * 1024 + (size_t)h_ * 128 + col;
            }
            if (OUT_F32) ((float*)Yv)[idx] = v;
            else         ((__bf16*)Yv)[idx] = (__bf16)v;
        }
    }
}

// ---------------------------------------------------------------------------
// Flash attention: 8 waves/block share one head; K/V chunks staged into LDS by
// the Tensor Data Mover (double-buffered, TENSORcnt), softmax online per wave.
//   Q: [32,1024,128]; K: [32,2048,128]; Vt: [32,128,2048]; out enh [32,1024,128]
// ---------------------------------------------------------------------------
__global__ __launch_bounds__(256) void attn_kernel(
        const __bf16* __restrict__ Q, const __bf16* __restrict__ K,
        const __bf16* __restrict__ Vt, __bf16* __restrict__ O) {
    __shared__ __bf16 ldsK[2][32 * 128];   // K chunk: 32 key-rows x 128 (8KB ea)
    __shared__ __bf16 ldsV[2][128 * 32];   // Vt chunk: 128 d-rows x 32 keys
    __shared__ __bf16 ldsP[8][16 * 32];    // per-wave P-tile bounce buffer

    int gw = (int)((blockIdx.x * blockDim.x + threadIdx.x) >> 5);  // 0..2047
    int wl = (int)(threadIdx.x >> 5);
    int bh = gw >> 6;       // 64 strips/head, 8 strips/block -> uniform per block
    int mt = gw & 63;
    int lane = threadIdx.x & 31;
    int r = lane & 15, hi = lane >> 4;

    const __bf16* Qh = Q  + (size_t)bh * 1024 * 128 + (size_t)mt * 16 * 128;
    const __bf16* Kh = K  + (size_t)bh * 2048 * 128;
    const __bf16* Vh = Vt + (size_t)bh * 128 * 2048;

    v16bf qa[4];
#pragma unroll
    for (int k = 0; k < 4; ++k) qa[k] = load_frag_rm(Qh + k * 32, 128);

    v8f o[8];
    float mrow[8], lrow[8];
#pragma unroll
    for (int t = 0; t < 8; ++t) {
#pragma unroll
        for (int j = 0; j < 8; ++j) o[t][j] = 0.f;
    }
#pragma unroll
    for (int j = 0; j < 8; ++j) { mrow[j] = -1e30f; lrow[j] = 0.f; }

    // Prefetch chunk 0 via TDM (wave 0 only; TENSORcnt tracks completion).
    if (wl == 0) {
        tdm_load_2d(Kh, ldsK[0], 128, 32, 128, 2048, 128);
        tdm_load_2d(Vh, ldsV[0], 32, 128, 2048, 128, 2048);
    }

    for (int n0 = 0; n0 < 2048; n0 += 32) {
        int buf = (n0 >> 5) & 1;
        if (wl == 0) {
            if (n0 + 32 < 2048) {  // overlap: issue next chunk before waiting
                tdm_load_2d(Kh + (size_t)(n0 + 32) * 128, ldsK[buf ^ 1],
                            128, 32, 128, 2048, 128);
                tdm_load_2d(Vh + (n0 + 32), ldsV[buf ^ 1],
                            32, 128, 2048, 128, 2048);
                __builtin_amdgcn_s_wait_tensorcnt(2);  // current pair complete
            } else {
                __builtin_amdgcn_s_wait_tensorcnt(0);
            }
        }
        __syncthreads();  // LDS chunk visible to all 8 waves

        // S = Q @ K^T for 32 keys: two 16-col tiles (B-frags from LDS)
        v8f s0, s1;
#pragma unroll
        for (int j = 0; j < 8; ++j) { s0[j] = 0.f; s1[j] = 0.f; }
#pragma unroll
        for (int k = 0; k < 4; ++k)
            s0 = wmma_bf16f32(qa[k], load_frag_rm(&ldsK[buf][0] + k * 32, 128), s0);
#pragma unroll
        for (int k = 0; k < 4; ++k)
            s1 = wmma_bf16f32(qa[k], load_frag_rm(&ldsK[buf][16 * 128] + k * 32, 128), s1);

        // scale + chunk row-max (reduce across 16-lane halves = C-layout rows)
        float cm[8];
#pragma unroll
        for (int j = 0; j < 8; ++j) {
            s0[j] *= SCALE_ATT; s1[j] *= SCALE_ATT;
            cm[j] = fmaxf(s0[j], s1[j]);
        }
#pragma unroll
        for (int d = 1; d < 16; d <<= 1)
#pragma unroll
            for (int j = 0; j < 8; ++j) cm[j] = fmaxf(cm[j], __shfl_xor(cm[j], d, 32));

        float rs[8];
#pragma unroll
        for (int j = 0; j < 8; ++j) {
            float nm = fmaxf(mrow[j], cm[j]);
            float corr = __expf(mrow[j] - nm);
            mrow[j] = nm;
            lrow[j] *= corr;
#pragma unroll
            for (int t = 0; t < 8; ++t) o[t][j] *= corr;
            s0[j] = __expf(s0[j] - nm);
            s1[j] = __expf(s1[j] - nm);
            rs[j] = s0[j] + s1[j];
        }
#pragma unroll
        for (int d = 1; d < 16; d <<= 1)
#pragma unroll
            for (int j = 0; j < 8; ++j) rs[j] += __shfl_xor(rs[j], d, 32);
#pragma unroll
        for (int j = 0; j < 8; ++j) lrow[j] += rs[j];

        // C-layout P tile -> LDS (row-major 16x32 bf16) -> A-layout fragment
        __bf16* pl = ldsP[wl];
#pragma unroll
        for (int j = 0; j < 8; ++j) {
            int row = j + hi * 8;
            pl[row * 32 + r]      = (__bf16)s0[j];
            pl[row * 32 + 16 + r] = (__bf16)s1[j];
        }
        asm volatile("s_wait_dscnt 0" ::: "memory");
        __builtin_amdgcn_wave_barrier();
        v16bf pa = load_frag_rm(pl, 32);

        // O += P @ V_chunk ; Vt chunk in LDS is [d,n] so B-frag cols contiguous
#pragma unroll
        for (int t = 0; t < 8; ++t) {
            v16bf vb = load_frag_rm(&ldsV[buf][(size_t)(t * 16) * 32], 32);
            o[t] = wmma_bf16f32(pa, vb, o[t]);
        }
        __syncthreads();  // all waves done with buf before TDM overwrites it
    }

    float inv[8];
#pragma unroll
    for (int j = 0; j < 8; ++j) inv[j] = 1.f / lrow[j];
#pragma unroll
    for (int t = 0; t < 8; ++t)
#pragma unroll
        for (int j = 0; j < 8; ++j) {
            int row = mt * 16 + j + hi * 8;
            int col = t * 16 + r;
            O[(size_t)bh * 1024 * 128 + (size_t)row * 128 + col] = (__bf16)(o[t][j] * inv[j]);
        }
}

// ---------------------------------------------------------------------------
extern "C" void kernel_launch(void* const* d_in, const int* in_sizes, int n_in,
                              void* d_out, int out_size, void* d_ws, size_t ws_size,
                              hipStream_t stream) {
    const float* query = (const float*)d_in[0];
    const float* fts   = (const float*)d_in[1];
    const float* wqc = (const float*)d_in[2];  const float* bqc = (const float*)d_in[3];
    const float* wq1 = (const float*)d_in[4];  const float* bq1 = (const float*)d_in[5];
    const float* wq2 = (const float*)d_in[6];  const float* bq2 = (const float*)d_in[7];
    const float* wk1 = (const float*)d_in[8];  const float* bk1 = (const float*)d_in[9];
    const float* wk2 = (const float*)d_in[10]; const float* bk2 = (const float*)d_in[11];
    const float* wv1 = (const float*)d_in[12]; const float* bv1 = (const float*)d_in[13];
    const float* wv2 = (const float*)d_in[14]; const float* bv2 = (const float*)d_in[15];
    const float* wo  = (const float*)d_in[16]; const float* bo  = (const float*)d_in[17];
    const float* wf1 = (const float*)d_in[18]; const float* bf1 = (const float*)d_in[19];
    const float* wf2 = (const float*)d_in[20]; const float* bf2 = (const float*)d_in[21];
    (void)in_sizes; (void)n_in; (void)out_size; (void)ws_size;

    char* ws = (char*)d_ws;
    size_t off = 0;
    auto alloc_bf = [&](size_t elems) -> __bf16* {
        __bf16* p = (__bf16*)(ws + off);
        off += ((elems * 2 + 255) & ~(size_t)255);
        return p;
    };

    __bf16* wqc_b = alloc_bf(1024 * 128);
    __bf16* wq1_b = alloc_bf(128 * 128);
    __bf16* wq2_b = alloc_bf(128 * 128);
    __bf16* wk1_b = alloc_bf(128 * 128);
    __bf16* wk2_b = alloc_bf(128 * 128);
    __bf16* wv1_b = alloc_bf(128 * 128);
    __bf16* wv2_b = alloc_bf(128 * 128);
    __bf16* wo_b  = alloc_bf(128 * 128);
    __bf16* wf1_b = alloc_bf(128 * 1024);
    __bf16* wf2_b = alloc_bf(128 * 128);
    __bf16* qT    = alloc_bf((size_t)4 * 1024 * 128);     // [B, hw, c]
    __bf16* ftT   = alloc_bf((size_t)32 * 2048 * 128);    // [bh, n, c]
    __bf16* qh    = alloc_bf((size_t)32 * 1024 * 128);    // after conv proj
    __bf16* tmp   = alloc_bf((size_t)32 * 2048 * 128);    // reused MLP hidden
    __bf16* qmat  = alloc_bf((size_t)32 * 1024 * 128);
    __bf16* kmat  = alloc_bf((size_t)32 * 2048 * 128);
    __bf16* vtm   = alloc_bf((size_t)32 * 128 * 2048);    // V transposed [bh,d,n]
    __bf16* enh   = alloc_bf((size_t)32 * 1024 * 128);
    __bf16* xbuf  = alloc_bf((size_t)4096 * 1024);        // [b*hw, h*128+o]
    __bf16* tbuf  = alloc_bf((size_t)4096 * 128);

    struct { const float* s; __bf16* d; int n; } wc[10] = {
        {wqc, wqc_b, 1024 * 128}, {wq1, wq1_b, 16384}, {wq2, wq2_b, 16384},
        {wk1, wk1_b, 16384}, {wk2, wk2_b, 16384}, {wv1, wv1_b, 16384},
        {wv2, wv2_b, 16384}, {wo, wo_b, 16384}, {wf1, wf1_b, 128 * 1024},
        {wf2, wf2_b, 16384}};
    for (int i = 0; i < 10; ++i)
        cast_f32_bf16<<<(wc[i].n + 255) / 256, 256, 0, stream>>>(wc[i].s, wc[i].d, wc[i].n);

    transpose_cast<<<dim3((128 * 1024 + 255) / 256, 4), 256, 0, stream>>>(query, qT, 128, 1024);
    transpose_cast<<<dim3((128 * 2048 + 255) / 256, 32), 256, 0, stream>>>(fts, ftT, 128, 2048);

    // q path: per-head conv proj, then Linear/LReLU/Linear/GELU
    gemm_bias_act<128, 0, 1, 0, false><<<256, 256, 0, stream>>>(qT, wqc_b, bqc, qh, 2048);
    gemm_bias_act<128, 1, 0, 0, false><<<256, 256, 0, stream>>>(qh, wq1_b, bq1, tmp, 2048);
    gemm_bias_act<128, 2, 0, 0, false><<<256, 256, 0, stream>>>(tmp, wq2_b, bq2, qmat, 2048);

    // k path
    gemm_bias_act<128, 1, 0, 0, false><<<512, 256, 0, stream>>>(ftT, wk1_b, bk1, tmp, 4096);
    gemm_bias_act<128, 2, 0, 0, false><<<512, 256, 0, stream>>>(tmp, wk2_b, bk2, kmat, 4096);

    // v path (second layer stores V^T per head for P@V B-frags)
    gemm_bias_act<128, 1, 0, 0, false><<<512, 256, 0, stream>>>(ftT, wv1_b, bv1, tmp, 4096);
    gemm_bias_act<128, 2, 0, 1, false><<<512, 256, 0, stream>>>(tmp, wv2_b, bv2, vtm, 4096);

    // attention (flash-style, TDM-staged K/V, online softmax)
    attn_kernel<<<256, 256, 0, stream>>>(qmat, kmat, vtm, enh);

    // out proj with head-interleaved store -> x [4096, 1024]
    gemm_bias_act<128, 0, 0, 2, false><<<256, 256, 0, stream>>>(enh, wo_b, bo, xbuf, 2048);

    // final MLP: K=1024 then K=128, LReLU both, last stores f32 to d_out
    gemm_bias_act<1024, 1, 0, 0, false><<<32, 256, 0, stream>>>(xbuf, wf1_b, bf1, tbuf, 256);
    gemm_bias_act<128, 1, 0, 0, true><<<32, 256, 0, stream>>>(tbuf, wf2_b, bf2, d_out, 256);
}
